// QASAModelAnalysis_57501022159103
// MI455X (gfx1250) — matrix operand order repaired and checked by reading, not compiled, not run on hardware
//
#include <hip/hip_runtime.h>
#include <math.h>

// ---------------------------------------------------------------------------
// Types for CDNA5 WMMA (wave32)
// ---------------------------------------------------------------------------
typedef __attribute__((ext_vector_type(16))) __bf16 v16b;
typedef __attribute__((ext_vector_type(8)))  __bf16 b8v;
typedef __attribute__((ext_vector_type(8)))  float  v8f;

#define HDIM 512
#define FFDIM 2048
#define NHEAD 4
#define DH 128
#define SEQ 128
#define BATCH 16
#define NTOK 2048   // BATCH*SEQ

// Async DMA global->LDS (CDNA5, tracked by ASYNCcnt). LDS operand is an
// addrspace(3) pointer = 32-bit wave-relative LDS address in one VGPR.
#define ASYNC_LOAD_B128(ldsptr, gptr)                                          \
    asm volatile("global_load_async_to_lds_b128 %0, %1, off"                   \
                 :: "v"((__attribute__((address_space(3))) char*)(ldsptr)),    \
                    "v"(gptr)                                                  \
                 : "memory")
#define WAIT_ASYNCCNT(n)                                                       \
    asm volatile("s_wait_asynccnt %0" :: "n"(n) : "memory")

// ---------------------------------------------------------------------------
// wave32 reductions
// ---------------------------------------------------------------------------
__device__ __forceinline__ float wave_sum(float v) {
    #pragma unroll
    for (int m = 16; m > 0; m >>= 1) v += __shfl_xor(v, m, 32);
    return v;
}
__device__ __forceinline__ float wave_max(float v) {
    #pragma unroll
    for (int m = 16; m > 0; m >>= 1) v = fmaxf(v, __shfl_xor(v, m, 32));
    return v;
}

// ---------------------------------------------------------------------------
// fp32 -> bf16 convert
// ---------------------------------------------------------------------------
__global__ void f2bf_kernel(const float* __restrict__ s, __bf16* __restrict__ d, int n) {
    int i = blockIdx.x * 256 + threadIdx.x;
    if (i < n) d[i] = (__bf16)s[i];
}

// ---------------------------------------------------------------------------
// Embedding + LayerNorm + positional encoding. One wave per token.
// ---------------------------------------------------------------------------
__global__ __launch_bounds__(256)
void embed_kernel(const float* __restrict__ x, const float* __restrict__ w,
                  const float* __restrict__ b, const float* __restrict__ g,
                  const float* __restrict__ lb, float* __restrict__ H,
                  __bf16* __restrict__ Hb) {
    int wave = threadIdx.x >> 5, lane = threadIdx.x & 31;
    int t = blockIdx.x * 8 + wave;
    float xv = x[t];
    float v[16];
    float s = 0.f;
    #pragma unroll
    for (int i = 0; i < 16; i++) { int d = lane + 32 * i; v[i] = xv * w[d] + b[d]; s += v[i]; }
    s = wave_sum(s);
    float mean = s * (1.0f / HDIM);
    float ss = 0.f;
    #pragma unroll
    for (int i = 0; i < 16; i++) { float dd = v[i] - mean; ss += dd * dd; }
    ss = wave_sum(ss);
    float inv = rsqrtf(ss * (1.0f / HDIM) + 1e-5f);
    float pos = (float)(t & (SEQ - 1));
    #pragma unroll
    for (int i = 0; i < 16; i++) {
        int d = lane + 32 * i;
        float o = (v[i] - mean) * inv * g[d] + lb[d];
        float ex = (float)(d & ~1);
        float freq = __expf(ex * (-9.210340371976184f / (float)HDIM));
        float ang = pos * freq;
        o += (d & 1) ? __cosf(ang) : __sinf(ang);
        H[(size_t)t * HDIM + d] = o;
        Hb[(size_t)t * HDIM + d] = (__bf16)o;
    }
}

// ---------------------------------------------------------------------------
// Residual + LayerNorm, writes fp32 + bf16 mirror. One wave per token.
// ---------------------------------------------------------------------------
__global__ __launch_bounds__(256)
void ln_residual_kernel(const float* __restrict__ X, const float* __restrict__ R,
                        const float* __restrict__ g, const float* __restrict__ b,
                        float* __restrict__ Y, __bf16* __restrict__ Yb) {
    int wave = threadIdx.x >> 5, lane = threadIdx.x & 31;
    int t = blockIdx.x * 8 + wave;
    const float* xr = X + (size_t)t * HDIM;
    const float* rr = R + (size_t)t * HDIM;
    float v[16];
    float s = 0.f;
    #pragma unroll
    for (int i = 0; i < 16; i++) { int d = lane + 32 * i; v[i] = xr[d] + rr[d]; s += v[i]; }
    s = wave_sum(s);
    float mean = s * (1.0f / HDIM);
    float ss = 0.f;
    #pragma unroll
    for (int i = 0; i < 16; i++) { float dd = v[i] - mean; ss += dd * dd; }
    ss = wave_sum(ss);
    float inv = rsqrtf(ss * (1.0f / HDIM) + 1e-5f);
    #pragma unroll
    for (int i = 0; i < 16; i++) {
        int d = lane + 32 * i;
        float o = (v[i] - mean) * inv * g[d] + b[d];
        Y[(size_t)t * HDIM + d] = o;
        Yb[(size_t)t * HDIM + d] = (__bf16)o;
    }
}

// ---------------------------------------------------------------------------
// Generic batched WMMA GEMM:  C[z] = act(A[z] (MxK) * W[z] (NxK)^T + bias)
// Double-buffered LDS staging via async DMA (ASYNCcnt), fragments read from
// LDS per the CDNA5 WMMA VGPR layouts. mode: 0=none, 1=relu, 2=gelu(exact)
// Block tile 64(M) x 128(N), K-step 32, 8 waves (wave = 16x64 tile).
// ---------------------------------------------------------------------------
__global__ __launch_bounds__(256)
void gemm_wmma_bf16(const __bf16* __restrict__ A, const __bf16* __restrict__ W,
                    const float* __restrict__ bias,
                    float* __restrict__ C, __bf16* __restrict__ Cb,
                    int M, int N, int K, int lda, int ldw, int ldc,
                    int zdiv, long long a1, long long a2,
                    long long w1, long long w2,
                    long long c1, long long c2, int mode) {
    __shared__ __bf16 Ash[2][64 * 32];    // 2 x 4KB
    __shared__ __bf16 Bsh[2][128 * 32];   // 2 x 8KB

    int z = blockIdx.z;
    A += (long long)(z / zdiv) * a1 + (long long)(z % zdiv) * a2;
    W += (long long)(z / zdiv) * w1 + (long long)(z % zdiv) * w2;
    long long zc = (long long)(z / zdiv) * c1 + (long long)(z % zdiv) * c2;

    int tid  = threadIdx.x;
    int lane = tid & 31;
    int wave = tid >> 5;
    int wm = wave & 3, wn = wave >> 2;
    int m0b = blockIdx.y * 64;
    int n0b = blockIdx.x * 128;
    int half = lane >> 4;   // lane half selects K sub-range per ISA layout
    int lr = lane & 15;

    // DMA assignments: A tile = 256 x 16B segs (1/thread); B tile = 512 (2/thread)
    int ar = tid >> 2, aseg = tid & 3;
    const __bf16* agbase = A + (long long)(m0b + ar) * lda + aseg * 8;
    int s1 = tid, s2 = tid + 256;
    int br1 = s1 >> 2, bg1 = s1 & 3;
    int br2 = s2 >> 2, bg2 = s2 & 3;
    const __bf16* wgbase1 = W + (long long)(n0b + br1) * ldw + bg1 * 8;
    const __bf16* wgbase2 = W + (long long)(n0b + br2) * ldw + bg2 * 8;
    __bf16* alds  = &Ash[0][ar * 32 + aseg * 8];
    __bf16* blds1 = &Bsh[0][br1 * 32 + bg1 * 8];
    __bf16* blds2 = &Bsh[0][br2 * 32 + bg2 * 8];

    #define ISSUE_TILE(buf, k0)                                        \
        ASYNC_LOAD_B128(alds  + (buf) * (64 * 32),  agbase  + (k0));   \
        ASYNC_LOAD_B128(blds1 + (buf) * (128 * 32), wgbase1 + (k0));   \
        ASYNC_LOAD_B128(blds2 + (buf) * (128 * 32), wgbase2 + (k0));

    v8f acc0 = {}, acc1 = {}, acc2 = {}, acc3 = {};
    ISSUE_TILE(0, 0)
    int cur = 0;
    for (int k0 = 0; k0 < K; k0 += 32) {
        if (k0 + 32 < K) {
            ISSUE_TILE(cur ^ 1, k0 + 32)
            WAIT_ASYNCCNT(3);   // drain current tile, keep prefetched in flight
        } else {
            WAIT_ASYNCCNT(0);
        }
        __syncthreads();

        // A fragment: row (wm*16+lr); k in [8h,8h+8) U [8h+16,8h+24)
        const __bf16* al = &Ash[cur][(wm * 16 + lr) * 32 + half * 8];
        union { v16b v; b8v h[2]; } ua;
        ua.h[0] = *(const b8v*)(al);
        ua.h[1] = *(const b8v*)(al + 16);
        union { v16b v; b8v h[2]; } uw;
#define WMMA_STEP(J, ACC)                                                          \
        {   const __bf16* bl = &Bsh[cur][(wn * 64 + 16 * (J) + lr) * 32 + half * 8]; \
            uw.h[0] = *(const b8v*)(bl);                                           \
            uw.h[1] = *(const b8v*)(bl + 16);                                      \
            ACC = __builtin_amdgcn_wmma_f32_16x16x32_bf16(false, ua.v, false, uw.v,\
                                                          (short)0, ACC, false, false); }
        WMMA_STEP(0, acc0)
        WMMA_STEP(1, acc1)
        WMMA_STEP(2, acc2)
        WMMA_STEP(3, acc3)
#undef WMMA_STEP
        __syncthreads();   // all waves done reading before buffer is overwritten
        cur ^= 1;
    }
    #undef ISSUE_TILE

    int m0 = m0b + wm * 16;
    int n0 = n0b + wn * 64;
    float* Cz = C + zc;
    __bf16* Cbz = Cb ? Cb + zc : (__bf16*)0;
#define EPILOG(J, ACC) {                                                          \
        int col = n0 + 16 * (J) + lr;                                             \
        float bv = bias ? bias[col] : 0.0f;                                       \
        for (int vv = 0; vv < 8; vv++) {                                          \
            int row = m0 + vv + half * 8;                                         \
            float val = ACC[vv] + bv;                                             \
            if (mode == 1) val = fmaxf(val, 0.0f);                                \
            else if (mode == 2) val = 0.5f * val * (1.0f + erff(val * 0.7071067811865475f)); \
            long long ci = (long long)row * ldc + col;                            \
            Cz[ci] = val;                                                         \
            if (Cbz) Cbz[ci] = (__bf16)val;                                       \
        } }
    EPILOG(0, acc0)
    EPILOG(1, acc1)
    EPILOG(2, acc2)
    EPILOG(3, acc3)
#undef EPILOG
}

// ---------------------------------------------------------------------------
// Split fused QKV into bf16 Q, K (token-major) and transposed V [b][h][d][s]
// ---------------------------------------------------------------------------
__global__ void split_qkv_kernel(const float* __restrict__ qkv, __bf16* __restrict__ qb,
                                 __bf16* __restrict__ kb, __bf16* __restrict__ vt) {
    int idx = blockIdx.x * 256 + threadIdx.x;    // over NTOK*HDIM
    int t = idx >> 9;
    int d = idx & (HDIM - 1);
    size_t base = (size_t)t * (3 * HDIM);
    qb[idx] = (__bf16)qkv[base + d];
    kb[idx] = (__bf16)qkv[base + HDIM + d];
    float v = qkv[base + 2 * HDIM + d];
    int b = t >> 7, s = t & (SEQ - 1);
    int h = d >> 7, dd = d & (DH - 1);
    vt[(((size_t)(b * NHEAD + h)) * DH + dd) * SEQ + s] = (__bf16)v;
}

// ---------------------------------------------------------------------------
// Row softmax over 128 keys (scaled). One wave per (b,h,q) row.
// ---------------------------------------------------------------------------
__global__ __launch_bounds__(256)
void softmax_kernel(const float* __restrict__ Sc, __bf16* __restrict__ P, float scale) {
    int wave = threadIdx.x >> 5, lane = threadIdx.x & 31;
    int row = blockIdx.x * 8 + wave;
    const float* p = Sc + (long long)row * SEQ;
    float v[4];
    float m = -1e30f;
    #pragma unroll
    for (int i = 0; i < 4; i++) { v[i] = p[lane + 32 * i] * scale; m = fmaxf(m, v[i]); }
    m = wave_max(m);
    float s = 0.f;
    #pragma unroll
    for (int i = 0; i < 4; i++) { v[i] = __expf(v[i] - m); s += v[i]; }
    s = wave_sum(s);
    float inv = 1.0f / s;
    __bf16* o = P + (long long)row * SEQ;
    #pragma unroll
    for (int i = 0; i < 4; i++) o[lane + 32 * i] = (__bf16)(v[i] * inv);
}

// ---------------------------------------------------------------------------
// q_in projection: xp[t][j] = tanh(h[t] . w[j] + b[j]),  j<8. One wave/token.
// ---------------------------------------------------------------------------
__global__ __launch_bounds__(256)
void qin_kernel(const float* __restrict__ H, const float* __restrict__ W,
                const float* __restrict__ B, float* __restrict__ XP) {
    int wave = threadIdx.x >> 5, lane = threadIdx.x & 31;
    int t = blockIdx.x * 8 + wave;
    float acc[8] = {0, 0, 0, 0, 0, 0, 0, 0};
    for (int d = lane; d < HDIM; d += 32) {
        float hv = H[(size_t)t * HDIM + d];
        #pragma unroll
        for (int j = 0; j < 8; j++) acc[j] += hv * W[j * HDIM + d];
    }
    #pragma unroll
    for (int j = 0; j < 8; j++) acc[j] = wave_sum(acc[j]);
    if (lane == 0) {
        #pragma unroll
        for (int j = 0; j < 8; j++) XP[t * 8 + j] = tanhf(acc[j] + B[j]);
    }
}

// LayerNorm over 8 + timestep add. One thread per token.
__global__ void qln_kernel(const float* __restrict__ XP, const float* __restrict__ g,
                           const float* __restrict__ b, float* __restrict__ XO, float tstep) {
    int t = blockIdx.x * 256 + threadIdx.x;
    if (t >= NTOK) return;
    float v[8];
    float s = 0.f;
    #pragma unroll
    for (int j = 0; j < 8; j++) { v[j] = XP[t * 8 + j]; s += v[j]; }
    float mean = s * 0.125f;
    float ss = 0.f;
    #pragma unroll
    for (int j = 0; j < 8; j++) { float dd = v[j] - mean; ss += dd * dd; }
    float inv = rsqrtf(ss * 0.125f + 1e-5f);
    #pragma unroll
    for (int j = 0; j < 8; j++) XO[t * 8 + j] = (v[j] - mean) * inv * g[j] + b[j] + tstep;
}

// ---------------------------------------------------------------------------
// 9-wire state-vector circuit, one wave per token, psi in LDS (4KB/wave).
// ---------------------------------------------------------------------------
__global__ __launch_bounds__(128)
void quantum_kernel(const float* __restrict__ XP, const float* __restrict__ QW,
                    float* __restrict__ Z) {
    __shared__ float2 psiAll[4][512];
    int wave = threadIdx.x >> 5, lane = threadIdx.x & 31;
    int t = blockIdx.x * 4 + wave;
    float2* psi = psiAll[wave];
    for (int i = lane; i < 512; i += 32) psi[i] = make_float2(0.f, 0.f);
    __syncthreads();
    if (lane == 0) psi[0] = make_float2(1.f, 0.f);
    __syncthreads();

    float ang[8];
    #pragma unroll
    for (int i = 0; i < 8; i++) ang[i] = XP[t * 8 + i];

    auto gate = [&](int w, float2 u00, float2 u01, float2 u10, float2 u11) {
        int pos = 8 - w;
        int stride = 1 << pos;
        for (int p = lane; p < 256; p += 32) {
            int i0 = ((p >> pos) << (pos + 1)) | (p & (stride - 1));
            int i1 = i0 | stride;
            float2 a = psi[i0], b = psi[i1];
            float2 r0 = make_float2(u00.x * a.x - u00.y * a.y + u01.x * b.x - u01.y * b.y,
                                    u00.x * a.y + u00.y * a.x + u01.x * b.y + u01.y * b.x);
            float2 r1 = make_float2(u10.x * a.x - u10.y * a.y + u11.x * b.x - u11.y * b.y,
                                    u10.x * a.y + u10.y * a.x + u11.x * b.y + u11.y * b.x);
            psi[i0] = r0; psi[i1] = r1;
        }
        __syncthreads();
    };
    auto rx = [&](int w, float th) {
        float c = __cosf(th * 0.5f), s = __sinf(th * 0.5f);
        gate(w, make_float2(c, 0.f), make_float2(0.f, -s), make_float2(0.f, -s), make_float2(c, 0.f));
    };
    auto rz = [&](int w, float th) {
        float c = __cosf(th * 0.5f), s = __sinf(th * 0.5f);
        gate(w, make_float2(c, -s), make_float2(0.f, 0.f), make_float2(0.f, 0.f), make_float2(c, s));
    };
    auto ry = [&](int w, float th) {
        float c = __cosf(th * 0.5f), s = __sinf(th * 0.5f);
        gate(w, make_float2(c, 0.f), make_float2(-s, 0.f), make_float2(s, 0.f), make_float2(c, 0.f));
    };
    auto cnot = [&](int c, int tt) {
        int pc = 8 - c, pt = 8 - tt;
        int hi = pc > pt ? pc : pt, lo = pc > pt ? pt : pc;
        for (int p = lane; p < 128; p += 32) {
            int xidx = p;
            xidx = ((xidx >> lo) << (lo + 1)) | (xidx & ((1 << lo) - 1));
            xidx = ((xidx >> hi) << (hi + 1)) | (xidx & ((1 << hi) - 1));
            int i0 = xidx | (1 << pc);
            int i1 = i0 | (1 << pt);
            float2 tmp = psi[i0]; psi[i0] = psi[i1]; psi[i1] = tmp;
        }
        __syncthreads();
    };

    for (int i = 0; i < 8; i++) { rx(i, ang[i]); rz(i, ang[i]); }
    for (int i = 0; i < 8; i++) { rx(i, QW[0 * 9 + i]); rz(i, QW[1 * 9 + i]); }
    for (int l = 1; l < 4; l++) {
        for (int i = 0; i < 8; i++) {
            cnot(i, (i + 1) & 7);
            ry(i, QW[l * 9 + i]);
            rz(i, QW[l * 9 + i]);
        }
        cnot(7, 8);
        ry(8, QW[l * 9 + 8]);
    }

    float zp[8] = {0, 0, 0, 0, 0, 0, 0, 0};
    for (int idx = lane; idx < 512; idx += 32) {
        float2 a = psi[idx];
        float pr = a.x * a.x + a.y * a.y;
        #pragma unroll
        for (int i = 0; i < 8; i++) zp[i] += ((idx >> (8 - i)) & 1) ? -pr : pr;
    }
    #pragma unroll
    for (int i = 0; i < 8; i++) zp[i] = wave_sum(zp[i]);
    if (lane == 0) {
        #pragma unroll
        for (int i = 0; i < 8; i++) Z[t * 8 + i] = zp[i];
    }
}

// q_out + residual: qout[t][d] = h[t][d] + b[d] + sum_j z[t][j]*W[d][j]
__global__ void qout_kernel(const float* __restrict__ H, const float* __restrict__ Z,
                            const float* __restrict__ W, const float* __restrict__ B,
                            float* __restrict__ Q, __bf16* __restrict__ Qb) {
    int idx = blockIdx.x * 256 + threadIdx.x;
    int t = idx >> 9;
    int d = idx & (HDIM - 1);
    float s = B[d];
    #pragma unroll
    for (int j = 0; j < 8; j++) s += Z[t * 8 + j] * W[d * 8 + j];
    float val = H[idx] + s;
    Q[idx] = val;
    Qb[idx] = (__bf16)val;
}

// final projection to scalar per token
__global__ __launch_bounds__(256)
void final_kernel(const float* __restrict__ H, const float* __restrict__ w,
                  const float* __restrict__ b, float* __restrict__ out) {
    int wave = threadIdx.x >> 5, lane = threadIdx.x & 31;
    int t = blockIdx.x * 8 + wave;
    float s = 0.f;
    for (int d = lane; d < HDIM; d += 32) s += H[(size_t)t * HDIM + d] * w[d];
    s = wave_sum(s);
    if (lane == 0) out[t] = s + b[0];
}

// ---------------------------------------------------------------------------
// Host orchestration
// ---------------------------------------------------------------------------
extern "C" void kernel_launch(void* const* d_in, const int* in_sizes, int n_in,
                              void* d_out, int out_size, void* d_ws, size_t ws_size,
                              hipStream_t stream) {
    auto IN = [&](int i) { return (const float*)d_in[i]; };

    // Parameter index map (setup_inputs insertion order):
    // 0:x 1:emb.w 2:emb.b 3:emb_ln.g 4:emb_ln.b
    // layers l in 0..2 at 5+12l: in_w,in_b,out_w,out_b,l1w,l1b,l2w,l2b,n1g,n1b,n2g,n2b
    // qlayer at 41: in_w,in_b,out_w,out_b, q_in.w,q_in.b,q_ln.g,q_ln.b,q_w,
    //               q_out.w,q_out.b, ffn1.w,ffn1.b, ffn2.w,ffn2.b, n1g,n1b,n2g,n2b
    // 60:out.w 61:out.b
    const int iw[4]  = {5, 17, 29, 41};
    const int ib_[4] = {6, 18, 30, 42};
    const int ow[4]  = {7, 19, 31, 43};
    const int ob[4]  = {8, 20, 32, 44};
    const int l1w[4] = {9, 21, 33, 52};
    const int l1b[4] = {10, 22, 34, 53};
    const int l2w[4] = {11, 23, 35, 54};
    const int l2b[4] = {12, 24, 36, 55};
    const int n1g[4] = {13, 25, 37, 56};
    const int n1b[4] = {14, 26, 38, 57};
    const int n2g[4] = {15, 27, 39, 58};
    const int n2b[4] = {16, 28, 40, 59};

    // Workspace carve
    char* wsp = (char*)d_ws;
    size_t off = 0;
    auto alloc = [&](size_t bytes) -> void* {
        void* p = wsp + off;
        off += (bytes + 255) & ~(size_t)255;
        return p;
    };

    __bf16 *Win[4], *Wout[4], *Wl1[4], *Wl2[4];
    for (int l = 0; l < 4; l++) {
        Win[l]  = (__bf16*)alloc((size_t)3 * HDIM * HDIM * 2);
        Wout[l] = (__bf16*)alloc((size_t)HDIM * HDIM * 2);
        Wl1[l]  = (__bf16*)alloc((size_t)FFDIM * HDIM * 2);
        Wl2[l]  = (__bf16*)alloc((size_t)HDIM * FFDIM * 2);
    }
    float*  h      = (float*)alloc((size_t)NTOK * HDIM * 4);
    __bf16* hb     = (__bf16*)alloc((size_t)NTOK * HDIM * 2);
    float*  qkv    = (float*)alloc((size_t)NTOK * 3 * HDIM * 4);
    __bf16* qb     = (__bf16*)alloc((size_t)NTOK * HDIM * 2);
    __bf16* kb     = (__bf16*)alloc((size_t)NTOK * HDIM * 2);
    __bf16* vt     = (__bf16*)alloc((size_t)NTOK * HDIM * 2);
    float*  scores = (float*)alloc((size_t)BATCH * NHEAD * SEQ * SEQ * 4);
    __bf16* pb     = (__bf16*)alloc((size_t)BATCH * NHEAD * SEQ * SEQ * 2);
    float*  attnO  = (float*)alloc((size_t)NTOK * HDIM * 4);
    __bf16* attnOb = (__bf16*)alloc((size_t)NTOK * HDIM * 2);
    float*  tmp    = (float*)alloc((size_t)NTOK * HDIM * 4);
    float*  ff1    = (float*)alloc((size_t)NTOK * FFDIM * 4);
    __bf16* ff1b   = (__bf16*)alloc((size_t)NTOK * FFDIM * 2);
    float*  ff2    = (float*)alloc((size_t)NTOK * HDIM * 4);
    float*  xp     = (float*)alloc((size_t)NTOK * 8 * 4);
    float*  xpn    = (float*)alloc((size_t)NTOK * 8 * 4);
    float*  zq     = (float*)alloc((size_t)NTOK * 8 * 4);
    float*  qres   = (float*)alloc((size_t)NTOK * HDIM * 4);
    __bf16* qresb  = (__bf16*)alloc((size_t)NTOK * HDIM * 2);

    auto conv = [&](const float* s, __bf16* d, int n) {
        f2bf_kernel<<<(n + 255) / 256, 256, 0, stream>>>(s, d, n);
    };
    for (int l = 0; l < 4; l++) {
        conv(IN(iw[l]),  Win[l],  3 * HDIM * HDIM);
        conv(IN(ow[l]),  Wout[l], HDIM * HDIM);
        conv(IN(l1w[l]), Wl1[l],  FFDIM * HDIM);
        conv(IN(l2w[l]), Wl2[l],  HDIM * FFDIM);
    }

    auto gemm = [&](const __bf16* A, const __bf16* W, const float* bias,
                    float* C, __bf16* Cb, int M, int N, int K,
                    int lda, int ldw, int ldc, int batch, int zdiv,
                    long long a1, long long a2, long long w1, long long w2,
                    long long c1, long long c2, int mode) {
        dim3 grid(N / 128, M / 64, batch);
        gemm_wmma_bf16<<<grid, 256, 0, stream>>>(A, W, bias, C, Cb, M, N, K, lda, ldw, ldc,
                                                 zdiv, a1, a2, w1, w2, c1, c2, mode);
    };

    embed_kernel<<<NTOK / 8, 256, 0, stream>>>(IN(0), IN(1), IN(2), IN(3), IN(4), h, hb);

    const float scale = 0.08838834764831845f;  // 1/sqrt(128)
    for (int l = 0; l < 4; l++) {
        // --- MHA ---
        gemm(hb, Win[l], IN(ib_[l]), qkv, nullptr, NTOK, 3 * HDIM, HDIM,
             HDIM, HDIM, 3 * HDIM, 1, 1, 0, 0, 0, 0, 0, 0, 0);
        split_qkv_kernel<<<NTOK * HDIM / 256, 256, 0, stream>>>(qkv, qb, kb, vt);
        // scores[z=b*4+h][q][k] = Q . K
        gemm(qb, kb, nullptr, scores, nullptr, SEQ, SEQ, DH,
             HDIM, HDIM, SEQ, BATCH * NHEAD, NHEAD,
             (long long)SEQ * HDIM, DH, (long long)SEQ * HDIM, DH,
             (long long)NHEAD * SEQ * SEQ, (long long)SEQ * SEQ, 0);
        softmax_kernel<<<BATCH * NHEAD * SEQ / 8, 256, 0, stream>>>(scores, pb, scale);
        // O[q][d] = P . V^T
        gemm(pb, vt, nullptr, attnO, attnOb, SEQ, DH, SEQ,
             SEQ, SEQ, HDIM, BATCH * NHEAD, NHEAD,
             (long long)NHEAD * SEQ * SEQ, (long long)SEQ * SEQ,
             (long long)NHEAD * DH * SEQ, (long long)DH * SEQ,
             (long long)SEQ * HDIM, DH, 0);
        gemm(attnOb, Wout[l], IN(ob[l]), tmp, nullptr, NTOK, HDIM, HDIM,
             HDIM, HDIM, HDIM, 1, 1, 0, 0, 0, 0, 0, 0, 0);
        ln_residual_kernel<<<NTOK / 8, 256, 0, stream>>>(h, tmp, IN(n1g[l]), IN(n1b[l]), h, hb);

        if (l < 3) {
            // --- standard FFN (ReLU) ---
            gemm(hb, Wl1[l], IN(l1b[l]), ff1, ff1b, NTOK, FFDIM, HDIM,
                 HDIM, HDIM, FFDIM, 1, 1, 0, 0, 0, 0, 0, 0, 1);
            gemm(ff1b, Wl2[l], IN(l2b[l]), ff2, nullptr, NTOK, HDIM, FFDIM,
                 FFDIM, FFDIM, HDIM, 1, 1, 0, 0, 0, 0, 0, 0, 0);
            ln_residual_kernel<<<NTOK / 8, 256, 0, stream>>>(h, ff2, IN(n2g[l]), IN(n2b[l]), h, hb);
        } else {
            // --- quantum block ---
            qin_kernel<<<NTOK / 8, 256, 0, stream>>>(h, IN(45), IN(46), xp);
            qln_kernel<<<NTOK / 256, 256, 0, stream>>>(xp, IN(47), IN(48), xpn, (float)SEQ);
            quantum_kernel<<<NTOK / 4, 128, 0, stream>>>(xpn, IN(49), zq);
            qout_kernel<<<NTOK * HDIM / 256, 256, 0, stream>>>(h, zq, IN(50), IN(51), qres, qresb);
            // FFN with exact GELU
            gemm(qresb, Wl1[3], IN(l1b[3]), ff1, ff1b, NTOK, FFDIM, HDIM,
                 HDIM, HDIM, FFDIM, 1, 1, 0, 0, 0, 0, 0, 0, 2);
            gemm(ff1b, Wl2[3], IN(l2b[3]), ff2, nullptr, NTOK, HDIM, FFDIM,
                 FFDIM, FFDIM, HDIM, 1, 1, 0, 0, 0, 0, 0, 0, 0);
            ln_residual_kernel<<<NTOK / 8, 256, 0, stream>>>(qres, ff2, IN(n2g[3]), IN(n2b[3]), h, hb);
        }
    }

    final_kernel<<<NTOK / 8, 256, 0, stream>>>(h, IN(60), IN(61), (float*)d_out);
}
